// TemporalInferenceBlock_38319698215415
// MI455X (gfx1250) — compile-verified, matrix operand order checked by reading, not run on hardware
//
#include <hip/hip_runtime.h>
#include <hip/hip_bf16.h>

// CDNA5 / gfx1250: wave32, WMMA 16x16x32 bf16 -> f32 accumulate.
// GEMM uses 2x2 register blocking: 32x32 output tile per wave,
// 4 v_wmma per k-step at 2 b128 loads per wmma.

typedef __attribute__((ext_vector_type(16))) __bf16 v16bf;
typedef __attribute__((ext_vector_type(8)))  float  v8f;

#define B_DIM  256
#define CIN    1024
#define COUT   1024
#define KTAPS  3
#define LCACHE 129
#define KD     (CIN * KTAPS)   // 3072 = reduction dim of both GEMMs

__device__ __forceinline__ unsigned short f2bf(float f) {
  // round-to-nearest-even fp32 -> bf16
  unsigned int u = __float_as_uint(f);
  u += 0x7fffu + ((u >> 16) & 1u);
  return (unsigned short)(u >> 16);
}

// ---------------------------------------------------------------------------
// s[o] = g[o] / sqrt(sum_{ik} v[o,i,k]^2)   (weight-norm scalar, folded into
// the GEMM epilogue instead of scaling the whole weight tensor)
// ---------------------------------------------------------------------------
__global__ __launch_bounds__(256) void scale_kernel(
    const float* __restrict__ v1, const float* __restrict__ g1,
    const float* __restrict__ v2, const float* __restrict__ g2,
    float* __restrict__ s1, float* __restrict__ s2) {
  const int o     = blockIdx.x & (COUT - 1);
  const int layer = blockIdx.x >> 10;
  const float* v = layer ? v2 : v1;
  const float* g = layer ? g2 : g1;
  float*       s = layer ? s2 : s1;

  const float* row = v + (size_t)o * KD;
  float acc = 0.f;
  for (int i = threadIdx.x; i < KD; i += 256) {
    float t = row[i];
    acc += t * t;
  }
  __shared__ float red[256];
  red[threadIdx.x] = acc;
  __syncthreads();
  for (int st = 128; st > 0; st >>= 1) {
    if (threadIdx.x < st) red[threadIdx.x] += red[threadIdx.x + st];
    __syncthreads();
  }
  if (threadIdx.x == 0) s[o] = g[o] / sqrtf(red[0]);
}

// ---------------------------------------------------------------------------
// fp32 weights -> bf16 (rows stay K-contiguous: exactly the B-operand layout)
// ---------------------------------------------------------------------------
__global__ __launch_bounds__(256) void convert_w(
    const float* __restrict__ v1, const float* __restrict__ v2,
    unsigned short* __restrict__ w1, unsigned short* __restrict__ w2) {
  const size_t n   = (size_t)COUT * KD;
  const size_t idx = (size_t)blockIdx.x * 256 + threadIdx.x;
  if (idx < n)          w1[idx]     = f2bf(v1[idx]);
  else if (idx < 2 * n) w2[idx - n] = f2bf(v2[idx - n]);
}

// ---------------------------------------------------------------------------
// Gather the 3 taps per (b, channel) row into packed bf16 A matrices.
// A1[b][3i+{0,1,2}] = {cache1[b,i,1], cache1[b,i,65], x[b,i]}
// A2[b][3i+{0,1}]   = {cache2[b,i,1], cache2[b,i,65]}   (slot 2 <- gemm1)
// ---------------------------------------------------------------------------
__global__ __launch_bounds__(256) void pack_taps(
    const float* __restrict__ x,
    const float* __restrict__ c1, const float* __restrict__ c2,
    unsigned short* __restrict__ A1, unsigned short* __restrict__ A2) {
  const int idx = blockIdx.x * 256 + threadIdx.x;       // b*1024 + i
  if (idx >= B_DIM * CIN) return;
  const int b = idx >> 10;
  const int i = idx & (CIN - 1);
  const float* r1 = c1 + (size_t)idx * LCACHE;
  const float* r2 = c2 + (size_t)idx * LCACHE;
  const size_t base = (size_t)b * KD + 3 * i;
  A1[base + 0] = f2bf(r1[1]);    // c1[:, :, ::64] tap 0  (post-shift)
  A1[base + 1] = f2bf(r1[65]);   // tap 1
  A1[base + 2] = f2bf(x[idx]);   // tap 2 = new sample
  A2[base + 0] = f2bf(r2[1]);
  A2[base + 1] = f2bf(r2[65]);
}

// ---------------------------------------------------------------------------
// One wave per 32x32 output tile (2x2 fragments of 16x16); 96 k-steps,
// 4 x v_wmma_f32_16x16x32_bf16 per step, 8 x b128 loads per step.
// Operand fragment layout (ISA 7.12.2, 16-bit A 16x32):
//   lane<16 : VGPR0-3 = K+0..7,  VGPR4-7 = K+16..23   (row = lane)
//   lane>=16: VGPR0-3 = K+8..15, VGPR4-7 = K+24..31
// Our B operand is W[o][k] (= B^T row-major) so it loads identically.
// ---------------------------------------------------------------------------
union Frag {
  v16bf v;
  uint4 u[2];
};

template <int MODE>
__global__ __launch_bounds__(256) void gemm_bf16_wmma(
    const unsigned short* __restrict__ A,    // [256][3072] bf16
    const unsigned short* __restrict__ W,    // [1024][3072] bf16
    const float* __restrict__ scale,         // [1024] weight-norm scalars
    const float* __restrict__ bias,          // [1024]
    unsigned short* __restrict__ outA2,      // MODE==1: write relu as tap 2
    const float* __restrict__ resid,         // MODE==2: x residual
    float* __restrict__ outF)                // MODE==2: final output
{
  const int lane = threadIdx.x & 31;
  const int tile = blockIdx.x * (256 >> 5) + (threadIdx.x >> 5); // 0..255
  const int mt   = tile >> 5;   // 8 blocks of M=256 (32 rows each)
  const int nt   = tile & 31;   // 32 blocks of N=1024 (32 cols each)
  const int half = lane >> 4;
  const int row  = lane & 15;

  const unsigned short* pa[2];
  const unsigned short* pb[2];
#pragma unroll
  for (int i = 0; i < 2; ++i) {
    pa[i] = A + (size_t)(mt * 32 + i * 16 + row) * KD + half * 8;
    pb[i] = W + (size_t)(nt * 32 + i * 16 + row) * KD + half * 8;
  }

  v8f acc[2][2];
#pragma unroll
  for (int i = 0; i < 2; ++i)
#pragma unroll
    for (int j = 0; j < 2; ++j)
      acc[i][j] = (v8f){0.f, 0.f, 0.f, 0.f, 0.f, 0.f, 0.f, 0.f};

#pragma unroll 2
  for (int k = 0; k < KD; k += 32) {
    Frag a[2], b[2];
#pragma unroll
    for (int i = 0; i < 2; ++i) {
      a[i].u[0] = *(const uint4*)(pa[i] + k);
      a[i].u[1] = *(const uint4*)(pa[i] + k + 16);
      b[i].u[0] = *(const uint4*)(pb[i] + k);
      b[i].u[1] = *(const uint4*)(pb[i] + k + 16);
    }
#pragma unroll
    for (int i = 0; i < 2; ++i)
#pragma unroll
      for (int j = 0; j < 2; ++j)
        acc[i][j] = __builtin_amdgcn_wmma_f32_16x16x32_bf16(
            /*neg_a=*/false, a[i].v, /*neg_b=*/false, b[j].v,
            /*c_mod=*/(short)0, acc[i][j],
            /*reuse_a=*/false, /*reuse_b=*/false);
  }

  // C/D layout: lane -> N, VGPR r -> M = r + 8*half
#pragma unroll
  for (int j = 0; j < 2; ++j) {
    const int   n  = nt * 32 + j * 16 + row;
    const float s  = scale[n];
    const float bi = bias[n];
#pragma unroll
    for (int i = 0; i < 2; ++i) {
#pragma unroll
      for (int r = 0; r < 8; ++r) {
        const int m = mt * 32 + i * 16 + half * 8 + r;
        float val = fmaxf(acc[i][j][r] * s + bi, 0.f);  // relu(wn-gemm + b)
        if (MODE == 1) {
          outA2[(size_t)m * KD + 3 * n + 2] = f2bf(val);  // tap 2, layer 2
        } else {
          val = fmaxf(val + resid[(size_t)m * COUT + n], 0.f);
          outF[(size_t)m * COUT + n] = val;
        }
      }
    }
  }
}

// ---------------------------------------------------------------------------
extern "C" void kernel_launch(void* const* d_in, const int* in_sizes, int n_in,
                              void* d_out, int out_size, void* d_ws,
                              size_t ws_size, hipStream_t stream) {
  const float* x  = (const float*)d_in[0];
  const float* c1 = (const float*)d_in[1];
  const float* c2 = (const float*)d_in[2];
  const float* v1 = (const float*)d_in[3];
  const float* g1 = (const float*)d_in[4];
  const float* b1 = (const float*)d_in[5];
  const float* v2 = (const float*)d_in[6];
  const float* g2 = (const float*)d_in[7];
  const float* b2 = (const float*)d_in[8];
  float* out = (float*)d_out;

  // workspace carve-up (~15.8 MB total; L2-resident on MI455X)
  char* ws = (char*)d_ws;
  size_t off = 0;
  unsigned short* W1 = (unsigned short*)(ws + off); off += (size_t)COUT * KD * 2;
  unsigned short* W2 = (unsigned short*)(ws + off); off += (size_t)COUT * KD * 2;
  unsigned short* A1 = (unsigned short*)(ws + off); off += (size_t)B_DIM * KD * 2;
  unsigned short* A2 = (unsigned short*)(ws + off); off += (size_t)B_DIM * KD * 2;
  float* s1 = (float*)(ws + off); off += COUT * 4;
  float* s2 = (float*)(ws + off); off += COUT * 4;
  (void)ws_size; (void)in_sizes; (void)n_in; (void)out_size;

  // 1) weight-norm scalars
  scale_kernel<<<2 * COUT, 256, 0, stream>>>(v1, g1, v2, g2, s1, s2);

  // 2) weights -> bf16
  {
    size_t total = 2ull * COUT * KD;
    convert_w<<<(unsigned)((total + 255) / 256), 256, 0, stream>>>(v1, v2, W1, W2);
  }

  // 3) tap gather -> packed bf16 activations
  pack_taps<<<(B_DIM * CIN + 255) / 256, 256, 0, stream>>>(x, c1, c2, A1, A2);

  // 4) layer 1 GEMM: out1 -> A2 tap-2 slots (bf16); 256 waves (32 blk x 8 wv)
  gemm_bf16_wmma<1><<<32, 256, 0, stream>>>(A1, W1, s1, b1, A2, nullptr, nullptr);

  // 5) layer 2 GEMM + residual + relu -> d_out (fp32)
  gemm_bf16_wmma<2><<<32, 256, 0, stream>>>(A2, W2, s2, b2, nullptr, x, out);
}
